// BinaryFinCast_84146999263308
// MI455X (gfx1250) — compile-verified, matrix operand order checked by reading, not run on hardware
//
#include <hip/hip_runtime.h>
#include <hip/hip_bf16.h>
#include <math.h>

// ---------------------------------------------------------------------------
// Model dims (compile-time)
//   B=16, S=2048, C=8, P=16 -> N=128 tokens/batch, M = B*N = 2048 rows
//   PD=128, IRH=512, D=512, NH=8, dh=64, L=12, E=4, H=2048
// ---------------------------------------------------------------------------

typedef __attribute__((ext_vector_type(16))) __bf16 v16bf;
typedef __attribute__((ext_vector_type(8)))  float  v8f;
typedef __attribute__((ext_vector_type(4)))  unsigned int v4u;
typedef __attribute__((ext_vector_type(4)))  int v4i_t;

union FragB { v16bf v; unsigned int u[8]; };

#ifndef __has_builtin
#define __has_builtin(x) 0
#endif

#if __has_builtin(__builtin_amdgcn_global_load_async_to_lds_b128)
#define USE_ASYNC_LDS 1
#else
#define USE_ASYNC_LDS 0
#endif

typedef __attribute__((address_space(1))) v4i_t glob_v4i;
typedef __attribute__((address_space(3))) v4i_t lds_v4i;

__device__ __forceinline__ glob_v4i* glob_v4(const void* p) {
  return (glob_v4i*)(unsigned long long)p;
}
__device__ __forceinline__ lds_v4i* lds_v4(const void* p) {
  // LDS flat addresses truncate to addr[31:0] (CDNA5 aperture rule)
  return (lds_v4i*)(unsigned int)(unsigned long long)p;
}
__device__ __forceinline__ void wait_async0() {
#if USE_ASYNC_LDS
#if __has_builtin(__builtin_amdgcn_s_wait_asynccnt)
  __builtin_amdgcn_s_wait_asynccnt(0);
#else
  asm volatile("s_wait_asynccnt 0" ::: "memory");
#endif
#endif
}

__device__ __forceinline__ unsigned short f2b(float f) {
  unsigned int u = __builtin_bit_cast(unsigned int, f);
  unsigned int r = u + 0x7fffu + ((u >> 16) & 1u);   // round-to-nearest-even
  return (unsigned short)(r >> 16);
}
__device__ __forceinline__ float b2f(unsigned short h) {
  unsigned int u = ((unsigned int)h) << 16;
  return __builtin_bit_cast(float, u);
}
__device__ __forceinline__ float gelu_exact(float x) {
  return 0.5f * x * (1.0f + erff(x * 0.7071067811865475f));
}

// Load a 16x32 bf16 WMMA fragment for one lane.
// row16 points at this lane's 32 contiguous bf16 (viewed as 16 uints).
// kb2 = (lane>>4)*4 : half-wave K split per CDNA5 ISA layout.
__device__ __forceinline__ v16bf load_frag(const unsigned int* row16, int kb2) {
  FragB f;
#pragma unroll
  for (int j = 0; j < 4; ++j) {
    f.u[j]     = row16[kb2 + j];       // K = kb*2 + {0..7}
    f.u[4 + j] = row16[kb2 + 8 + j];   // K = kb*2 + {16..23}
  }
  return f.v;
}

// ---------------------------------------------------------------------------
// Front end: patchify + instance-norm + residual MLP + p2m + freq embedding
// one block (128 thr) per row (B*N = 2048 rows); writes h fp32 [2048,512]
// ---------------------------------------------------------------------------
__global__ __launch_bounds__(128) void front_kernel(
    const float* __restrict__ x, const int* __restrict__ freq_id,
    const float* __restrict__ irw1, const float* __restrict__ irb1,
    const float* __restrict__ irw2, const float* __restrict__ irb2,
    const float* __restrict__ p2mw, const float* __restrict__ p2mb,
    const float* __restrict__ femb, float* __restrict__ h)
{
  int row = blockIdx.x;            // b*128 + n
  int tid = threadIdx.x;           // 0..127
  int b = row >> 7;
  __shared__ float red[128];
  __shared__ float patch[128];
  __shared__ float hid[512];
  __shared__ float rr[128];

  const float* px = x + (long)row * 128;   // patch is 128 contiguous floats
  float xv = px[tid];
  red[tid] = xv; __syncthreads();
  for (int s = 64; s; s >>= 1) { if (tid < s) red[tid] += red[tid + s]; __syncthreads(); }
  float mu = red[0] / 128.f; __syncthreads();
  float d = xv - mu;
  red[tid] = d * d; __syncthreads();
  for (int s = 64; s; s >>= 1) { if (tid < s) red[tid] += red[tid + s]; __syncthreads(); }
  float rstd = rsqrtf(red[0] / 128.f + 1e-6f);
  __syncthreads();
  patch[tid] = d * rstd;
  __syncthreads();

  for (int o = tid; o < 512; o += 128) {
    float a = irb1[o];
    for (int k = 0; k < 128; ++k) a += patch[k] * irw1[k * 512 + o];
    hid[o] = gelu_exact(a);
  }
  __syncthreads();
  {
    float a = irb2[tid];
    for (int k = 0; k < 512; ++k) a += hid[k] * irw2[k * 128 + tid];
    rr[tid] = patch[tid] + a;
  }
  __syncthreads();
  int fe = freq_id[b];
  for (int o = tid; o < 512; o += 128) {
    float a = p2mb[o];
    for (int k = 0; k < 128; ++k) a += rr[k] * p2mw[k * 512 + o];
    h[(long)row * 512 + o] = a + femb[fe * 512 + o];
  }
}

// ---------------------------------------------------------------------------
// LayerNorm (eps 1e-5): fp32 in -> bf16 out. One block (256 thr) per row.
// ---------------------------------------------------------------------------
__global__ __launch_bounds__(256) void ln_kernel(
    const float* __restrict__ src, const float* __restrict__ g,
    const float* __restrict__ bta, unsigned short* __restrict__ dst)
{
  int row = blockIdx.x, tid = threadIdx.x;
  __shared__ float red[256];
  const float* p = src + (long)row * 512;
  float a0 = p[tid], a1 = p[tid + 256];
  red[tid] = a0 + a1; __syncthreads();
  for (int s = 128; s; s >>= 1) { if (tid < s) red[tid] += red[tid + s]; __syncthreads(); }
  float mu = red[0] / 512.f; __syncthreads();
  float d0 = a0 - mu, d1 = a1 - mu;
  red[tid] = d0 * d0 + d1 * d1; __syncthreads();
  for (int s = 128; s; s >>= 1) { if (tid < s) red[tid] += red[tid + s]; __syncthreads(); }
  float rstd = rsqrtf(red[0] / 512.f + 1e-5f);
  dst[(long)row * 512 + tid]       = f2b(g[tid] * d0 * rstd + bta[tid]);
  dst[(long)row * 512 + tid + 256] = f2b(g[tid + 256] * d1 * rstd + bta[tid + 256]);
}

// ---------------------------------------------------------------------------
// Tiled bf16 WMMA GEMM:  out = epilogue(A[M,K]bf16 @ W[K,N]f32 + bias)
// Block 256 thr = 8 waves. Macro tile 128(M) x 64(N), K-step 32.
// A tile staged via GLOBAL_LOAD_ASYNC_TO_LDS_B128 (ASYNCcnt) when available;
// W tile converted fp32->bf16 on the VALU into transposed LDS.
// MODE 0: bias -> bf16 out            (QKV)
// MODE 1: bias, outF += val           (attn out projection residual)
// MODE 2: bias, exact GELU -> bf16    (expert up-proj)
// MODE 3: bias, outF += rowscale[m*4]*val  (expert down-proj, gate-scaled)
// ---------------------------------------------------------------------------
template <int MODE>
__global__ __launch_bounds__(256) void gemm_bf16_kernel(
    const unsigned short* __restrict__ A, const float* __restrict__ W,
    const float* __restrict__ bias, float* __restrict__ outF,
    unsigned short* __restrict__ outB, const float* __restrict__ rowscale,
    int M, int Kd, int N, long zsA, long zsW, long zsB, long zsOut)
{
  __shared__ unsigned short As[128 * 32];   // A tile  [m][k] bf16   (8 KB)
  __shared__ unsigned short Ws[64 * 32];    // W tile  [n][k] bf16^T (4 KB)

  int z = blockIdx.z;
  A += (long)z * zsA; W += (long)z * zsW; bias += (long)z * zsB;
  if (MODE == 0 || MODE == 2) outB += (long)z * zsOut; else outF += (long)z * zsOut;

  int tid = threadIdx.x;
  int lane = tid & 31, wave = tid >> 5;
  int wr = wave >> 1, wc = wave & 1;
  int m0 = blockIdx.y * 128, n0 = blockIdx.x * 64;
  int l15 = lane & 15;
  int kb2 = (lane >> 4) * 4;

  v8f acc[2][2] = {};
  int ksteps = Kd >> 5;
  for (int kk = 0; kk < ksteps; ++kk) {
    int k0 = kk * 32;
    // ---- A tile: 128x32 bf16 = 512 x 16B chunks, 2 per thread ----
#if USE_ASYNC_LDS
#pragma unroll
    for (int i = 0; i < 2; ++i) {
      int s = tid + i * 256;               // 0..511
      int r = s >> 2, c = s & 3;
      const unsigned short* gp = A + (long)(m0 + r) * Kd + k0 + c * 8;
      __builtin_amdgcn_global_load_async_to_lds_b128(
          glob_v4(gp), lds_v4(As + s * 8), 0, 0);
    }
#else
#pragma unroll
    for (int i = 0; i < 2; ++i) {
      int s = tid + i * 256;
      int r = s >> 2, c = s & 3;
      const v4u* gp = (const v4u*)(A + (long)(m0 + r) * Kd + k0);
      ((v4u*)As)[s] = gp[c];
    }
#endif
    // ---- W tile: 32x64 fp32 -> bf16 transposed [n][k] ----
#pragma unroll
    for (int i = 0; i < 8; ++i) {
      int s = tid + i * 256;               // 2048 floats = 32x64
      int kr = s >> 6, nc = s & 63;
      float f = W[(long)(k0 + kr) * N + (n0 + nc)];
      Ws[nc * 32 + kr] = f2b(f);
    }
    // prefetch next K-slice of W into cache (global_prefetch_b8)
    if (kk + 1 < ksteps) {
      int pr = tid >> 3, pc = (tid & 7) << 3;
      __builtin_prefetch(W + (long)(k0 + 32 + pr) * N + n0 + pc, 0, 3);
    }
    wait_async0();
    __syncthreads();

    v16bf afr[2], bfr[2];
#pragma unroll
    for (int i = 0; i < 2; ++i) {
      int ar = wr * 32 + i * 16 + l15;
      afr[i] = load_frag((const unsigned int*)(As + ar * 32), kb2);
    }
#pragma unroll
    for (int j = 0; j < 2; ++j) {
      int bc = wc * 32 + j * 16 + l15;
      bfr[j] = load_frag((const unsigned int*)(Ws + bc * 32), kb2);
    }
#pragma unroll
    for (int i = 0; i < 2; ++i)
#pragma unroll
      for (int j = 0; j < 2; ++j)
        acc[i][j] = __builtin_amdgcn_wmma_f32_16x16x32_bf16(
            false, afr[i], false, bfr[j], (short)0, acc[i][j], false, false);
    __syncthreads();
  }

#pragma unroll
  for (int i = 0; i < 2; ++i) {
#pragma unroll
    for (int j = 0; j < 2; ++j) {
#pragma unroll
      for (int r = 0; r < 8; ++r) {
        int m = m0 + wr * 32 + i * 16 + (lane >> 4) * 8 + r;
        int n = n0 + wc * 32 + j * 16 + l15;
        float val = acc[i][j][r] + bias[n];
        long idx = (long)m * N + n;
        if (MODE == 0) {
          outB[idx] = f2b(val);
        } else if (MODE == 2) {
          outB[idx] = f2b(gelu_exact(val));
        } else if (MODE == 1) {
          outF[idx] += val;
        } else {
          outF[idx] += rowscale[(long)m * 4] * val;   // rowscale pre-offset by expert
        }
      }
    }
  }
}

// ---------------------------------------------------------------------------
// Attention: per (batch, head, query-half). Block 256 thr = 8 waves.
// Waves 0..3 handle cols 0..63, waves 4..7 cols 64..127 (16 q-rows each).
// S fp32 in LDS, causal softmax, then P@V via WMMA (P packed fp32->bf16).
// ---------------------------------------------------------------------------
__global__ __launch_bounds__(256) void attn_kernel(
    const unsigned short* __restrict__ qkv,      // [2048,1536] bf16 (q|k|v)
    unsigned short* __restrict__ attn_out)       // [2048,512]  bf16
{
  __shared__ float Sld[64 * 128];                // 32 KB scores (this q-half)
  __shared__ unsigned short vT[64 * 128];        // 16 KB v transposed [d][n]
  int bx = blockIdx.x;
  int b = bx >> 4;
  int r2 = bx & 15;
  int hd = r2 >> 1;
  int half = r2 & 1;
  int base_m = b * 128;
  int qbase = half * 64;
  int tid = threadIdx.x, lane = tid & 31, wave = tid >> 5;
  int wr = wave & 3, wcH = wave >> 2;
  int l15 = lane & 15;
  int kb2 = (lane >> 4) * 4;

  for (int i = 0; i < 32; ++i) {                 // stage vT
    int s = tid + i * 256;                       // 0..8191
    int n = s >> 6, dc = s & 63;
    vT[dc * 128 + n] = qkv[(long)(base_m + n) * 1536 + 1024 + hd * 64 + dc];
  }
  __syncthreads();

  {                                              // S = (q @ k^T) / 8
    int qr = qbase + wr * 16 + l15;
#pragma unroll
    for (int cb = 0; cb < 4; ++cb) {
      int coln = wcH * 64 + cb * 16 + l15;
      v8f acc = {};
#pragma unroll
      for (int ks = 0; ks < 2; ++ks) {
        v16bf afr = load_frag((const unsigned int*)(qkv + (long)(base_m + qr) * 1536 + hd * 64 + ks * 32), kb2);
        v16bf bfr = load_frag((const unsigned int*)(qkv + (long)(base_m + coln) * 1536 + 512 + hd * 64 + ks * 32), kb2);
        acc = __builtin_amdgcn_wmma_f32_16x16x32_bf16(false, afr, false, bfr, (short)0, acc, false, false);
      }
#pragma unroll
      for (int r = 0; r < 8; ++r) {
        int rl = wr * 16 + (lane >> 4) * 8 + r;
        int cl = wcH * 64 + cb * 16 + l15;
        Sld[rl * 128 + cl] = acc[r] * 0.125f;
      }
    }
  }
  __syncthreads();

  if (tid < 64) {                                // causal softmax per row
    int row = tid;
    int qn = qbase + row;
    float* sr = Sld + row * 128;
    float mx = -3.0e38f;
    for (int c = 0; c <= qn; ++c) mx = fmaxf(mx, sr[c]);
    float sum = 0.f;
    for (int c = 0; c <= qn; ++c) { float e = expf(sr[c] - mx); sr[c] = e; sum += e; }
    float inv = 1.f / sum;
    for (int c = 0; c <= qn; ++c) sr[c] *= inv;
    for (int c = qn + 1; c < 128; ++c) sr[c] = 0.f;
  }
  __syncthreads();

  {                                              // O = P @ V
    int prow = wr * 16 + l15;
    const float* pr = Sld + prow * 128;
    int kb = (lane >> 4) * 8;
#pragma unroll
    for (int j = 0; j < 2; ++j) {
      int dcol = wcH * 32 + j * 16 + l15;
      v8f acc = {};
#pragma unroll
      for (int ks = 0; ks < 4; ++ks) {
        FragB a;
        const float* pk = pr + ks * 32;
#pragma unroll
        for (int jj = 0; jj < 4; ++jj) {
          unsigned int lo = f2b(pk[kb + 2 * jj]);
          unsigned int hi = f2b(pk[kb + 2 * jj + 1]);
          a.u[jj] = lo | (hi << 16);
          lo = f2b(pk[kb + 16 + 2 * jj]);
          hi = f2b(pk[kb + 16 + 2 * jj + 1]);
          a.u[4 + jj] = lo | (hi << 16);
        }
        v16bf bfr = load_frag((const unsigned int*)(vT + dcol * 128 + ks * 32), kb2);
        acc = __builtin_amdgcn_wmma_f32_16x16x32_bf16(false, a.v, false, bfr, (short)0, acc, false, false);
      }
#pragma unroll
      for (int r = 0; r < 8; ++r) {
        int qn = qbase + wr * 16 + (lane >> 4) * 8 + r;
        int d = wcH * 32 + j * 16 + l15;
        attn_out[(long)(base_m + qn) * 512 + hd * 64 + d] = f2b(acc[r]);
      }
    }
  }
}

// ---------------------------------------------------------------------------
// Gate: softmax over 4 experts + top-2 mask -> combine weights [M,4]
// ---------------------------------------------------------------------------
__global__ __launch_bounds__(256) void gate_kernel(
    const unsigned short* __restrict__ hn, const float* __restrict__ gw,
    const float* __restrict__ gb, float* __restrict__ wgt)
{
  int m = blockIdx.x * 256 + threadIdx.x;
  float a[4] = {gb[0], gb[1], gb[2], gb[3]};
  const unsigned short* hp = hn + (long)m * 512;
  for (int k = 0; k < 512; ++k) {
    float hv = b2f(hp[k]);
    const float* wr = gw + k * 4;
    a[0] += hv * wr[0]; a[1] += hv * wr[1]; a[2] += hv * wr[2]; a[3] += hv * wr[3];
  }
  float mx = fmaxf(fmaxf(a[0], a[1]), fmaxf(a[2], a[3]));
  float e[4], s = 0.f;
  for (int i = 0; i < 4; ++i) { e[i] = expf(a[i] - mx); s += e[i]; }
  for (int i = 0; i < 4; ++i) e[i] /= s;
  int i0 = 0;
  for (int i = 1; i < 4; ++i) if (e[i] > e[i0]) i0 = i;
  int i1 = -1;
  for (int i = 0; i < 4; ++i) { if (i == i0) continue; if (i1 < 0 || e[i] > e[i1]) i1 = i; }
  for (int i = 0; i < 4; ++i) wgt[(long)m * 4 + i] = (i == i0 || i == i1) ? e[i] : 0.f;
}

// ---------------------------------------------------------------------------
// Head: final LN, last-token LN, dot with head_w, sigmoid. One block / batch.
// ---------------------------------------------------------------------------
__global__ __launch_bounds__(128) void final_kernel(
    const float* __restrict__ h, const float* __restrict__ fng, const float* __restrict__ fnb,
    const float* __restrict__ hg, const float* __restrict__ hb,
    const float* __restrict__ hw, const float* __restrict__ hbias, float* __restrict__ out)
{
  int b = blockIdx.x, tid = threadIdx.x;
  __shared__ float red[128];
  __shared__ float y[512];
  const float* row = h + ((long)(b * 128 + 127)) * 512;
  float s = 0.f;
  for (int i = tid; i < 512; i += 128) s += row[i];
  red[tid] = s; __syncthreads();
  for (int st = 64; st; st >>= 1) { if (tid < st) red[tid] += red[tid + st]; __syncthreads(); }
  float mu = red[0] / 512.f; __syncthreads();
  s = 0.f;
  for (int i = tid; i < 512; i += 128) { float d = row[i] - mu; s += d * d; }
  red[tid] = s; __syncthreads();
  for (int st = 64; st; st >>= 1) { if (tid < st) red[tid] += red[tid + st]; __syncthreads(); }
  float rstd = rsqrtf(red[0] / 512.f + 1e-5f); __syncthreads();
  for (int i = tid; i < 512; i += 128) y[i] = fng[i] * (row[i] - mu) * rstd + fnb[i];
  __syncthreads();
  s = 0.f;
  for (int i = tid; i < 512; i += 128) s += y[i];
  red[tid] = s; __syncthreads();
  for (int st = 64; st; st >>= 1) { if (tid < st) red[tid] += red[tid + st]; __syncthreads(); }
  float mu2 = red[0] / 512.f; __syncthreads();
  s = 0.f;
  for (int i = tid; i < 512; i += 128) { float d = y[i] - mu2; s += d * d; }
  red[tid] = s; __syncthreads();
  for (int st = 64; st; st >>= 1) { if (tid < st) red[tid] += red[tid + st]; __syncthreads(); }
  float rstd2 = rsqrtf(red[0] / 512.f + 1e-5f); __syncthreads();
  s = 0.f;
  for (int i = tid; i < 512; i += 128) {
    float z = hg[i] * (y[i] - mu2) * rstd2 + hb[i];
    s += z * hw[i];
  }
  red[tid] = s; __syncthreads();
  for (int st = 64; st; st >>= 1) { if (tid < st) red[tid] += red[tid + st]; __syncthreads(); }
  if (tid == 0) {
    float logit = red[0] + hbias[0];
    out[b] = logit;
    out[16 + b] = 1.f / (1.f + expf(-logit));
  }
}

// ---------------------------------------------------------------------------
extern "C" void kernel_launch(void* const* d_in, const int* in_sizes, int n_in,
                              void* d_out, int out_size, void* d_ws, size_t ws_size,
                              hipStream_t stream)
{
  (void)in_sizes; (void)n_in; (void)out_size; (void)ws_size;
  const float* x        = (const float*)d_in[0];
  const int*   freq_id  = (const int*)d_in[1];
  const float* ir_w1    = (const float*)d_in[2];
  const float* ir_b1    = (const float*)d_in[3];
  const float* ir_w2    = (const float*)d_in[4];
  const float* ir_b2    = (const float*)d_in[5];
  const float* p2m_w    = (const float*)d_in[6];
  const float* p2m_b    = (const float*)d_in[7];
  const float* femb     = (const float*)d_in[8];
  const float* ln1_g    = (const float*)d_in[9];
  const float* ln1_b    = (const float*)d_in[10];
  const float* qkv_w    = (const float*)d_in[11];
  const float* qkv_b    = (const float*)d_in[12];
  const float* out_w    = (const float*)d_in[13];
  const float* out_b    = (const float*)d_in[14];
  const float* ln2_g    = (const float*)d_in[15];
  const float* ln2_b    = (const float*)d_in[16];
  const float* gate_w   = (const float*)d_in[17];
  const float* gate_b   = (const float*)d_in[18];
  const float* exp_w1   = (const float*)d_in[19];
  const float* exp_b1   = (const float*)d_in[20];
  const float* exp_w2   = (const float*)d_in[21];
  const float* exp_b2   = (const float*)d_in[22];
  const float* fn_g     = (const float*)d_in[23];
  const float* fn_b     = (const float*)d_in[24];
  const float* head_g   = (const float*)d_in[25];
  const float* head_b   = (const float*)d_in[26];
  const float* head_w   = (const float*)d_in[27];
  const float* head_bias= (const float*)d_in[28];
  float* out = (float*)d_out;

  char* ws = (char*)d_ws;
  float*          h    = (float*)(ws);                        // 4 MB  [2048,512] f32
  unsigned short* hn   = (unsigned short*)(ws + (4L  << 20)); // 2 MB  [2048,512] bf16
  unsigned short* qkvB = (unsigned short*)(ws + (6L  << 20)); // 6 MB  [2048,1536] bf16
  unsigned short* aO   = (unsigned short*)(ws + (12L << 20)); // 2 MB  [2048,512] bf16
  float*          wgt  = (float*)(ws + (14L << 20));          // 32 KB [2048,4] f32
  unsigned short* hid  = (unsigned short*)(ws + (15L << 20)); // 32 MB [4,2048,2048] bf16

  front_kernel<<<2048, 128, 0, stream>>>(x, freq_id, ir_w1, ir_b1, ir_w2, ir_b2,
                                         p2m_w, p2m_b, femb, h);
  for (int l = 0; l < 12; ++l) {
    ln_kernel<<<2048, 256, 0, stream>>>(h, ln1_g + l * 512, ln1_b + l * 512, hn);
    gemm_bf16_kernel<0><<<dim3(24, 16, 1), 256, 0, stream>>>(
        hn, qkv_w + (long)l * 512 * 1536, qkv_b + l * 1536,
        nullptr, qkvB, nullptr, 2048, 512, 1536, 0, 0, 0, 0);
    attn_kernel<<<256, 256, 0, stream>>>(qkvB, aO);
    gemm_bf16_kernel<1><<<dim3(8, 16, 1), 256, 0, stream>>>(
        aO, out_w + (long)l * 512 * 512, out_b + l * 512,
        h, nullptr, nullptr, 2048, 512, 512, 0, 0, 0, 0);
    ln_kernel<<<2048, 256, 0, stream>>>(h, ln2_g + l * 512, ln2_b + l * 512, hn);
    gate_kernel<<<8, 256, 0, stream>>>(hn, gate_w + l * 512 * 4, gate_b + l * 4, wgt);
    gemm_bf16_kernel<2><<<dim3(32, 16, 4), 256, 0, stream>>>(
        hn, exp_w1 + (long)l * 4 * 512 * 2048, exp_b1 + (long)l * 4 * 2048,
        nullptr, hid, nullptr, 2048, 512, 2048,
        0, (long)512 * 2048, 2048, (long)2048 * 2048);
    for (int e = 0; e < 4; ++e) {
      gemm_bf16_kernel<3><<<dim3(8, 16, 1), 256, 0, stream>>>(
          hid + (long)e * 2048 * 2048,
          exp_w2 + ((long)l * 4 + e) * 2048 * 512,
          exp_b2 + ((long)l * 4 + e) * 512,
          h, nullptr, wgt + e, 2048, 2048, 512, 0, 0, 0, 0);
    }
  }
  final_kernel<<<16, 128, 0, stream>>>(h, fn_g, fn_b, head_g, head_b,
                                       head_w, head_bias, out);
}
